// SelfProjectionDev_48241072668799
// MI455X (gfx1250) — matrix operand order repaired and checked by reading, not compiled, not run on hardware
//
#include <hip/hip_runtime.h>
#include <hip/hip_bf16.h>

typedef __bf16 bf16;
typedef __bf16 v8bf  __attribute__((ext_vector_type(8)));
typedef __bf16 v16bf __attribute__((ext_vector_type(16)));
typedef float  v8f   __attribute__((ext_vector_type(8)));

#define B_    128
#define H_    512
#define W_    512
#define P_    64
#define DEP_  4
#define K1_   512
#define EPS_  1e-5f
#define SCALE_ 64.0f   /* (H/P)*(W/P) */

// ---------------------------------------------------------------------------
// WMMA fragment helpers (bf16 A/B, f32 accum), CDNA5 16x16x32 layouts.
// A (16x32, row-major src, pitch in elems): lane r=lane&15 owns row m0+r.
//   lanes<16: K = k0+0..7 and k0+16..23 ; lanes>=16: K = k0+8..15 and k0+24..31
// B (32x16 from packed W[n][k], pitch in elems): lane owns col n0+(lane&15);
//   lanes<16: K = k0+0..7 and k0+16..23 ; lanes>=16: K = k0+8..15 and k0+24..31
// C/D: vgpr g -> row m0 + 8*(lane>>4) + g, col n0 + (lane&15)
// ---------------------------------------------------------------------------
__device__ __forceinline__ v16bf load_frag_a(const bf16* base, int pitch,
                                             int m0, int k0, int lane) {
  const int r = lane & 15, half = lane >> 4;
  const bf16* p0 = base + (size_t)(m0 + r) * pitch + k0 + half * 8;
  v8bf lo = *(const v8bf*)(p0);
  v8bf hi = *(const v8bf*)(p0 + 16);
  v16bf a;
#pragma unroll
  for (int i = 0; i < 8; ++i) { a[i] = lo[i]; a[i + 8] = hi[i]; }
  return a;
}

__device__ __forceinline__ v16bf load_frag_b(const bf16* base, int pitch,
                                             int n0, int k0, int lane) {
  const int c = lane & 15, half = lane >> 4;
  const bf16* p0 = base + (size_t)(n0 + c) * pitch + k0 + half * 8;
  v8bf lo = *(const v8bf*)(p0);
  v8bf hi = *(const v8bf*)(p0 + 16);
  v16bf b;
#pragma unroll
  for (int i = 0; i < 8; ++i) { b[i] = lo[i]; b[i + 8] = hi[i]; }
  return b;
}

__device__ __forceinline__ v8f wmma_bf16(v16bf a, v16bf b, v8f c) {
  return __builtin_amdgcn_wmma_f32_16x16x32_bf16(
      /*neg_a=*/false, a, /*neg_b=*/false, b,
      /*c_mod=*/(short)0, c, /*reuse_a=*/false, /*reuse_b=*/false);
}

// numerically-stable (max, sumexp) merge
__device__ __forceinline__ void smerge(float& m, float& s, float m2, float s2) {
  float mm = fmaxf(m, m2);
  s = s * __expf(m - mm) + s2 * __expf(m2 - mm);
  m = mm;
}

// per-C-tile (max, sumexp) over this lane's 8 rows, merged with the partner
// lane (xor 16, same column other row-half), then merged into running (m,s)
__device__ __forceinline__ void tile_softmax_update(const v8f& acc,
                                                    float& mrun, float& srun) {
  float tm = acc[0];
#pragma unroll
  for (int g = 1; g < 8; ++g) tm = fmaxf(tm, acc[g]);
  float ts = 0.f;
#pragma unroll
  for (int g = 0; g < 8; ++g) ts += __expf(acc[g] - tm);
  float pm = __shfl_xor(tm, 16, 32);
  float ps = __shfl_xor(ts, 16, 32);
  float mm = tm, ss = ts;
  smerge(mm, ss, pm, ps);
  smerge(mrun, srun, mm, ss);
}

// cross-wave flat-softmax column reduce: 8 wave-partials per column ->
// out[p] = colsum_exp[p] / total_sum  (== softmax(...).sum(rows)[p])
__device__ __forceinline__ void colreduce(float (*rm)[64], float (*rs)[64],
                                          float* colm, float* cols,
                                          float* stmp, float* out, int tid) {
  if (tid < 64) {
    float m = -1e30f, s = 0.f;
#pragma unroll
    for (int w = 0; w < 8; ++w) smerge(m, s, rm[w][tid], rs[w][tid]);
    colm[tid] = m; cols[tid] = s;
  }
  __syncthreads();
  if (tid == 0) {
    float M = -1e30f;
    for (int p = 0; p < 64; ++p) M = fmaxf(M, colm[p]);
    float S = 0.f;
    for (int p = 0; p < 64; ++p) S += cols[p] * __expf(colm[p] - M);
    stmp[0] = M; stmp[1] = S;
  }
  __syncthreads();
  if (tid < 64) out[tid] = cols[tid] * __expf(colm[tid] - stmp[0]) / stmp[1];
  __syncthreads();
}

// ---------------------------------------------------------------------------
// Kernel 1: per-batch scaled mean / unbiased std of x
// ---------------------------------------------------------------------------
__global__ __launch_bounds__(256) void stats_kernel(const float* __restrict__ x,
                                                    float* __restrict__ meanstd) {
  __shared__ float r1[256], r2[256];
  const int b = blockIdx.x, tid = threadIdx.x;
  const float* xb = x + (size_t)b * H_ * W_;
  float s = 0.f, q = 0.f;
  for (int i = tid; i < H_ * W_; i += 256) { float v = xb[i]; s += v; q += v * v; }
  r1[tid] = s; r2[tid] = q; __syncthreads();
  for (int st = 128; st > 0; st >>= 1) {
    if (tid < st) { r1[tid] += r1[tid + st]; r2[tid] += r2[tid + st]; }
    __syncthreads();
  }
  if (tid == 0) {
    const float n = (float)(H_ * W_);
    float mean = r1[0] / n;
    float var  = (r2[0] - n * mean * mean) / (n - 1.0f);
    meanstd[b * 2 + 0] = mean * SCALE_;
    meanstd[b * 2 + 1] = sqrtf(var) * SCALE_;
  }
}

// ---------------------------------------------------------------------------
// Kernel 2: cast x -> bf16, both orientations (LDS-tiled transpose)
// ---------------------------------------------------------------------------
__global__ __launch_bounds__(256) void pack_x_kernel(const float* __restrict__ x,
                                                     bf16* __restrict__ xb,
                                                     bf16* __restrict__ xtb) {
  __shared__ float tile[32][33];
  const int bx = blockIdx.x, by = blockIdx.y, b = blockIdx.z;
  const int tx = threadIdx.x, ty = threadIdx.y;  // block (32,8)
  const float* xs = x + (size_t)b * H_ * W_;
  bf16* xo  = xb  + (size_t)b * H_ * W_;
  bf16* xto = xtb + (size_t)b * H_ * W_;
#pragma unroll
  for (int j = 0; j < 4; ++j) {
    int h = by * 32 + ty + j * 8, w = bx * 32 + tx;
    float v = xs[(size_t)h * W_ + w];
    xo[(size_t)h * W_ + w] = (bf16)v;
    tile[ty + j * 8][tx] = v;
  }
  __syncthreads();
#pragma unroll
  for (int j = 0; j < 4; ++j) {
    int wr = bx * 32 + ty + j * 8;   // row of x^T
    int hc = by * 32 + tx;           // col of x^T
    xto[(size_t)wr * H_ + hc] = (bf16)tile[tx][ty + j * 8];
  }
}

// ---------------------------------------------------------------------------
// Kernel 3: pack a weight tensor [DEP][K][P_] f32 -> [DEP][P_][K] bf16
// ---------------------------------------------------------------------------
__global__ __launch_bounds__(256) void pack_w_kernel(const float* __restrict__ src,
                                                     bf16* __restrict__ dst, int K) {
  const int idx = blockIdx.x * 256 + threadIdx.x;
  const int total = DEP_ * K * P_;
  if (idx >= total) return;
  const int d = idx / (K * P_), rem = idx % (K * P_);
  const int k = rem / P_, p = rem % P_;
  dst[((size_t)d * P_ + p) * K + k] = (bf16)src[idx];
}

// ---------------------------------------------------------------------------
// Kernel 4: per (b, depth, side) fused WMMA pipeline.
//   Stage1: Y = x_side @ [o_rel | o]  ([512,128]), 2x2 register blocking
//           (two M-tiles x {rel,proj} weights -> 4 independent WMMA chains),
//           online flat softmax of the rel half; Yproj^T stored bf16 in LDS.
//   Stage2: Pt = Yproj^T @ p_w  ([64,64], K=512), 2 N-tiles share A fragment
//   Stage3: Prel = Pt @ p_rel_w ([64,64], K=64) + flat softmax
//   F[p] = sqrt(orel_sum[p] / prel_sum[p])
// ---------------------------------------------------------------------------
__global__ __launch_bounds__(256) void proj_stage_kernel(
    const bf16* __restrict__ xb,   const bf16* __restrict__ xtb,
    const bf16* __restrict__ orelT, const bf16* __restrict__ oT,
    const bf16* __restrict__ pT,    const bf16* __restrict__ prelT,
    float* __restrict__ PtOut,      float* __restrict__ FOut) {
  __shared__ bf16 Yt[P_][K1_ + 8];     // Yproj^T (row p, K = original rows)
  __shared__ bf16 Ptb[P_][P_ + 8];     // Pt as bf16 A-matrix for stage 3
  __shared__ float red_m[8][64], red_s[8][64];
  __shared__ float colm[64], cols[64];
  __shared__ float orel_s[64], prel_s[64];
  __shared__ float stmp[2];

  const int tid  = threadIdx.x;
  const int lane = tid & 31;
  const int wave = tid >> 5;

  const int blk  = blockIdx.x;
  const int side = blk & 1;
  const int d    = (blk >> 1) & 3;
  const int b    = blk >> 3;

  const bf16* xs = (side ? xtb : xb) + (size_t)b * H_ * W_;
  const size_t wofs = (size_t)(side * DEP_ + d) * P_;
  const bf16* orelW = orelT + wofs * K1_;
  const bf16* oW    = oT    + wofs * K1_;
  const bf16* pW    = pT    + wofs * K1_;
  const bf16* prW   = prelT + wofs * P_;
  float* ptg = PtOut + (size_t)((b * DEP_ + d) * 2 + side) * (P_ * P_);
  float* fg  = FOut  + (size_t)((b * DEP_ + d) * 2 + side) * P_;

  // warm the small per-(d,side) weight blocks into cache (global_prefetch_b8)
  {
    const char* w0 = (const char*)orelW;
    const char* w1 = (const char*)oW;
    const char* w2 = (const char*)pW;
    for (size_t off = (size_t)tid * 64; off < (size_t)P_ * K1_ * 2;
         off += 256 * 64) {
      __builtin_prefetch(w0 + off, 0, 1);
      __builtin_prefetch(w1 + off, 0, 1);
      __builtin_prefetch(w2 + off, 0, 1);
    }
  }

  // ---------------- Stage 1 ----------------
  for (int ntp = 0; ntp < 4; ++ntp) {
    const int n0 = ntp * 16;
    float mrun = -1e30f, srun = 0.f;
#pragma unroll
    for (int mtp = 0; mtp < 2; ++mtp) {
      const int m0 = wave * 64 + mtp * 32;     // tiles at m0 and m0+16
      v8f aR0 = {}, aR1 = {}, aP0 = {}, aP1 = {};
#pragma unroll 4
      for (int k0 = 0; k0 < K1_; k0 += 32) {
        v16bf a0 = load_frag_a(xs, W_, m0,      k0, lane);
        v16bf a1 = load_frag_a(xs, W_, m0 + 16, k0, lane);
        v16bf bR = load_frag_b(orelW, K1_, n0, k0, lane);
        v16bf bP = load_frag_b(oW,    K1_, n0, k0, lane);
        aR0 = wmma_bf16(a0, bR, aR0);
        aR1 = wmma_bf16(a1, bR, aR1);
        aP0 = wmma_bf16(a0, bP, aP0);
        aP1 = wmma_bf16(a1, bP, aP1);
      }
      // rel half: online flat softmax
      tile_softmax_update(aR0, mrun, srun);
      tile_softmax_update(aR1, mrun, srun);
      // proj half: Y^T into LDS (bf16), contiguous-K rows
      const int pcol = n0 + (lane & 15);
      const int kb0  = m0 + (lane >> 4) * 8;
      v8bf ov0, ov1;
#pragma unroll
      for (int g = 0; g < 8; ++g) { ov0[g] = (bf16)aP0[g]; ov1[g] = (bf16)aP1[g]; }
      *(v8bf*)&Yt[pcol][kb0]      = ov0;
      *(v8bf*)&Yt[pcol][kb0 + 16] = ov1;
    }
    if (lane < 16) {
      red_m[wave][n0 + lane] = mrun;
      red_s[wave][n0 + lane] = srun;
    }
  }
  __syncthreads();
  colreduce(red_m, red_s, colm, cols, stmp, orel_s, tid);

  // ---------------- Stage 2: Pt = Yt @ pW (K=512) ----------------
  {
    const int t0 = wave * 2;
    const int mt = t0 >> 2, ntA = t0 & 3;      // pair (mt,ntA),(mt,ntA+1)
    const int m0 = mt * 16, n0A = ntA * 16, n0B = n0A + 16;
    v8f acc0 = {}, acc1 = {};
#pragma unroll 4
    for (int k0 = 0; k0 < K1_; k0 += 32) {
      v16bf a  = load_frag_a(&Yt[0][0], K1_ + 8, m0, k0, lane);
      v16bf b0 = load_frag_b(pW, K1_, n0A, k0, lane);
      v16bf b1 = load_frag_b(pW, K1_, n0B, k0, lane);
      acc0 = wmma_bf16(a, b0, acc0);
      acc1 = wmma_bf16(a, b1, acc1);
    }
    const int c = lane & 15, half = lane >> 4;
#pragma unroll
    for (int g = 0; g < 8; ++g) {
      const int m = m0 + half * 8 + g;
      ptg[m * 64 + n0A + c] = acc0[g];
      ptg[m * 64 + n0B + c] = acc1[g];
      Ptb[m][n0A + c] = (bf16)acc0[g];
      Ptb[m][n0B + c] = (bf16)acc1[g];
    }
  }
  // init stage-3 reduction slots (not every wave covers every column)
  for (int i = tid; i < 8 * 64; i += 256) {
    (&red_m[0][0])[i] = -1e30f;
    (&red_s[0][0])[i] = 0.f;
  }
  __syncthreads();

  // ---------------- Stage 3: Prel = Ptb @ prW (K=64), flat softmax ----------
  {
    const int t0 = wave * 2;
    const int mt = t0 >> 2, ntA = t0 & 3;
    const int m0 = mt * 16, n0A = ntA * 16, n0B = n0A + 16;
    v8f acc0 = {}, acc1 = {};
#pragma unroll
    for (int k0 = 0; k0 < P_; k0 += 32) {
      v16bf a  = load_frag_a(&Ptb[0][0], P_ + 8, m0, k0, lane);
      v16bf b0 = load_frag_b(prW, P_, n0A, k0, lane);
      v16bf b1 = load_frag_b(prW, P_, n0B, k0, lane);
      acc0 = wmma_bf16(a, b0, acc0);
      acc1 = wmma_bf16(a, b1, acc1);
    }
    float m0r = -1e30f, s0r = 0.f, m1r = -1e30f, s1r = 0.f;
    tile_softmax_update(acc0, m0r, s0r);
    tile_softmax_update(acc1, m1r, s1r);
    if (lane < 16) {
      red_m[wave][n0A + lane] = m0r;  red_s[wave][n0A + lane] = s0r;
      red_m[wave][n0B + lane] = m1r;  red_s[wave][n0B + lane] = s1r;
    }
  }
  __syncthreads();
  colreduce(red_m, red_s, colm, cols, stmp, prel_s, tid);

  if (tid < 64) fg[tid] = sqrtf(orel_s[tid] / prel_s[tid]);
}

// ---------------------------------------------------------------------------
// Kernel 5: per batch — combine sides, renormalize, accumulate depths.
// ---------------------------------------------------------------------------
__global__ __launch_bounds__(256) void combine_kernel(
    const float* __restrict__ Pt, const float* __restrict__ F,
    const float* __restrict__ meanstd, float* __restrict__ out) {
  __shared__ float red[256], red2[256], Fjs[64], Fis[64], sh[2];
  const int tid = threadIdx.x, b = blockIdx.x;
  const float xm = meanstd[b * 2 + 0];
  const float xs = meanstd[b * 2 + 1];
  float acc[16];
#pragma unroll
  for (int i = 0; i < 16; ++i) acc[i] = 0.f;

  for (int d = 0; d < DEP_; ++d) {
    const float* Ptj = Pt + (size_t)((b * DEP_ + d) * 2 + 0) * (P_ * P_);
    const float* Pti = Pt + (size_t)((b * DEP_ + d) * 2 + 1) * (P_ * P_);
    if (tid < 64) {
      Fjs[tid] = F[((b * DEP_ + d) * 2 + 0) * P_ + tid];
      Fis[tid] = F[((b * DEP_ + d) * 2 + 1) * P_ + tid];
    }
    __syncthreads();
    float v[16], s = 0.f, q = 0.f;
#pragma unroll
    for (int i = 0; i < 16; ++i) {
      const int e = tid + 256 * i;
      const int m = e >> 6, n = e & 63;
      float val = Ptj[e] * Fjs[m] + Pti[n * 64 + m] * Fis[n];
      v[i] = val; s += val; q += val * val;
    }
    red[tid] = s; red2[tid] = q; __syncthreads();
    for (int st = 128; st > 0; st >>= 1) {
      if (tid < st) { red[tid] += red[tid + st]; red2[tid] += red2[tid + st]; }
      __syncthreads();
    }
    if (tid == 0) {
      float mean = red[0] * (1.0f / 4096.0f);
      float var  = (red2[0] - 4096.0f * mean * mean) * (1.0f / 4095.0f);
      sh[0] = mean;
      sh[1] = 1.0f / (sqrtf(var) + EPS_);
    }
    __syncthreads();
#pragma unroll
    for (int i = 0; i < 16; ++i) acc[i] += (v[i] - sh[0]) * sh[1] * xs + xm;
    __syncthreads();
  }
#pragma unroll
  for (int i = 0; i < 16; ++i) out[(size_t)b * 4096 + tid + 256 * i] = acc[i];
}

// ---------------------------------------------------------------------------
// Host launcher
// ---------------------------------------------------------------------------
extern "C" void kernel_launch(void* const* d_in, const int* in_sizes, int n_in,
                              void* d_out, int out_size, void* d_ws, size_t ws_size,
                              hipStream_t stream) {
  const float* x        = (const float*)d_in[0];
  const float* o_xj     = (const float*)d_in[1];
  const float* o_xi     = (const float*)d_in[2];
  const float* p_xj     = (const float*)d_in[3];
  const float* p_xi     = (const float*)d_in[4];
  const float* o_rel_xj = (const float*)d_in[5];
  const float* o_rel_xi = (const float*)d_in[6];
  const float* p_rel_xj = (const float*)d_in[7];
  const float* p_rel_xi = (const float*)d_in[8];

  // workspace layout (bytes)
  constexpr size_t SZ_XB    = (size_t)B_ * H_ * W_ * 2;       // 64 MiB
  constexpr size_t SZ_WT    = (size_t)2 * DEP_ * P_ * K1_ * 2;// 512 KiB each
  constexpr size_t SZ_PRELT = (size_t)2 * DEP_ * P_ * P_ * 2; // 64 KiB
  constexpr size_t OFF_XB    = 0;
  constexpr size_t OFF_XTB   = OFF_XB + SZ_XB;
  constexpr size_t OFF_ORELT = OFF_XTB + SZ_XB;
  constexpr size_t OFF_OT    = OFF_ORELT + SZ_WT;
  constexpr size_t OFF_PT    = OFF_OT + SZ_WT;
  constexpr size_t OFF_PRELT = OFF_PT + SZ_WT;
  constexpr size_t OFF_MS    = OFF_PRELT + SZ_PRELT;
  constexpr size_t OFF_PTOUT = OFF_MS + 1024;
  constexpr size_t OFF_F     = OFF_PTOUT + (size_t)B_ * DEP_ * 2 * P_ * P_ * 4;

  char* ws = (char*)d_ws;
  bf16*  xb      = (bf16*)(ws + OFF_XB);
  bf16*  xtb     = (bf16*)(ws + OFF_XTB);
  bf16*  orelT   = (bf16*)(ws + OFF_ORELT);
  bf16*  oT      = (bf16*)(ws + OFF_OT);
  bf16*  pT      = (bf16*)(ws + OFF_PT);
  bf16*  prelT   = (bf16*)(ws + OFF_PRELT);
  float* meanstd = (float*)(ws + OFF_MS);
  float* PtBuf   = (float*)(ws + OFF_PTOUT);
  float* FBuf    = (float*)(ws + OFF_F);

  // 1) stats + bf16 packing
  stats_kernel<<<B_, 256, 0, stream>>>(x, meanstd);
  pack_x_kernel<<<dim3(W_ / 32, H_ / 32, B_), dim3(32, 8), 0, stream>>>(x, xb, xtb);

  const int wb512 = (DEP_ * K1_ * P_ + 255) / 256;
  const int wb64  = (DEP_ * P_ * P_ + 255) / 256;
  pack_w_kernel<<<wb512, 256, 0, stream>>>(o_rel_xj, orelT,                     K1_);
  pack_w_kernel<<<wb512, 256, 0, stream>>>(o_rel_xi, orelT + (size_t)DEP_ * P_ * K1_, K1_);
  pack_w_kernel<<<wb512, 256, 0, stream>>>(o_xj,     oT,                        K1_);
  pack_w_kernel<<<wb512, 256, 0, stream>>>(o_xi,     oT    + (size_t)DEP_ * P_ * K1_, K1_);
  pack_w_kernel<<<wb512, 256, 0, stream>>>(p_xj,     pT,                        K1_);
  pack_w_kernel<<<wb512, 256, 0, stream>>>(p_xi,     pT    + (size_t)DEP_ * P_ * K1_, K1_);
  pack_w_kernel<<<wb64,  256, 0, stream>>>(p_rel_xj, prelT,                     P_);
  pack_w_kernel<<<wb64,  256, 0, stream>>>(p_rel_xi, prelT + (size_t)DEP_ * P_ * P_,  P_);

  // 2) heavy WMMA stage: one workgroup per (b, depth, side)
  proj_stage_kernel<<<B_ * DEP_ * 2, 256, 0, stream>>>(
      xb, xtb, orelT, oT, pT, prelT, PtBuf, FBuf);

  // 3) combine + renormalize + depth accumulate
  combine_kernel<<<B_, 256, 0, stream>>>(PtBuf, FBuf, meanstd, (float*)d_out);

  (void)in_sizes; (void)n_in; (void)out_size; (void)ws_size;
}
